// Attention_5841155523069
// MI455X (gfx1250) — matrix operand order repaired and checked by reading, not compiled, not run on hardware
//
#include <hip/hip_runtime.h>
#include <stdint.h>

// ---------------------------------------------------------------------------
// Problem constants (from reference): T=2048, d_model=2048, H=16, d_h=128
// ---------------------------------------------------------------------------
#define T_SEQ   2048
#define D_MODEL 2048
#define N_HEAD  16
#define D_HEAD  128
#define QKV_N   (3 * D_MODEL)           // 6144
#define NEG_INF (-1.0e9f)
#define ATTN_SCALE 0.08838834764831845f // 1/sqrt(128)

typedef __attribute__((ext_vector_type(16))) __bf16 v16bf;
typedef __attribute__((ext_vector_type(8)))  __bf16 v8bf;
typedef __attribute__((ext_vector_type(8)))  float  v8f;
// GCC-style int4 matching the async-builtin's parameter type
// ('__attribute__((__vector_size__(4 * sizeof(int)))) int' per hipcc).
typedef int gv4i __attribute__((vector_size(16)));

// Async global->LDS path (CDNA5): guarded so the file compiles on either
// toolchain; histogram shows whether the async path lowered.
#if __has_builtin(__builtin_amdgcn_global_load_async_to_lds_b128) && \
    __has_builtin(__builtin_amdgcn_s_wait_asynccnt)
#define USE_ASYNC_LDS 1
#else
#define USE_ASYNC_LDS 0
#endif

// A-fragment (16x32 bf16, wave32): lane l holds row (l&15); element e maps to
// k = e + 16*(l>>4). Elements 0..7 and 8..15 are each
// k-contiguous -> two 16-byte loads.
union AFrag { v16bf v; v8bf h[2]; };

__device__ inline v8f wmma_bf16(const AFrag& a, v16bf b, v8f c) {
    return __builtin_amdgcn_wmma_f32_16x16x32_bf16(
        /*neg_a=*/false, a.v, /*neg_b=*/false, b,
        /*c_mod=*/(short)0, c, /*reuse_a=*/false, /*reuse_b=*/false);
}

// ---------------------------------------------------------------------------
// 1) RMSNorm: h_bf16[t][c] = bf16( ln_w[c] * x[t][c] * rsqrt(mean(x^2)+eps) )
// ---------------------------------------------------------------------------
__global__ __launch_bounds__(256)
void rmsnorm_bf16_kernel(const float* __restrict__ x,
                         const float* __restrict__ w,
                         __bf16* __restrict__ h) {
    __shared__ float red[8];
    __shared__ float rres;
    const int row = blockIdx.x;
    const float* xr = x + (size_t)row * D_MODEL;
    float ss = 0.f;
    for (int i = threadIdx.x; i < D_MODEL; i += blockDim.x) {
        float v = xr[i]; ss += v * v;
    }
    for (int m = 16; m > 0; m >>= 1) ss += __shfl_xor(ss, m, 32);
    if ((threadIdx.x & 31) == 0) red[threadIdx.x >> 5] = ss;
    __syncthreads();
    if (threadIdx.x == 0) {
        float t = 0.f;
        for (int i = 0; i < 8; ++i) t += red[i];
        rres = rsqrtf(t / (float)D_MODEL + 1e-6f);
    }
    __syncthreads();
    const float r = rres;
    for (int i = threadIdx.x; i < D_MODEL; i += blockDim.x)
        h[(size_t)row * D_MODEL + i] = (__bf16)(w[i] * xr[i] * r);
}

// ---------------------------------------------------------------------------
// 2) f32 -> bf16 conversion (weights)
// ---------------------------------------------------------------------------
__global__ __launch_bounds__(256)
void cvt_bf16_kernel(const float* __restrict__ src, __bf16* __restrict__ dst,
                     int n) {
    for (int i = blockIdx.x * blockDim.x + threadIdx.x; i < n;
         i += gridDim.x * blockDim.x)
        dst[i] = (__bf16)src[i];
}

// ---------------------------------------------------------------------------
// 3) GEMM:  C[M][N] (f32) = A[M][K]bf16 @ W[N][K]bf16 ^T   (+ epilogue)
//    Block = 256 threads (8 waves, 2Mx4N), block tile 64(M) x 256(N).
//    W k-slice (256 rows x 32 k = 16 KB) staged in LDS, double buffered,
//    via GLOBAL_LOAD_ASYNC_TO_LDS_B128 (ASYNCcnt) when available.
//    Each wave: 32x64 tile = 8 accumulators, 8 WMMA per 32-k step.
//    mode 0: C = A@W^T                  (QKV projection)
//    mode 1: C = x + mask[r]*(A@W^T)    (output projection + residual)
// ---------------------------------------------------------------------------
#define GEMM_BM 64
#define GEMM_BN 256
#define GEMM_BK 32

__device__ __forceinline__
void gemm_stage_w(__bf16* dst, const __bf16* __restrict__ W,
                  int nBase, int K, int kk, int srow, int schunk) {
#pragma unroll
    for (int it = 0; it < 4; ++it) {
        const int row = srow + 64 * it;
        const __bf16* g = W + (size_t)(nBase + row) * K + kk + schunk * 8;
        __bf16* l = dst + row * GEMM_BK + schunk * 8;
#if USE_ASYNC_LDS
        __builtin_amdgcn_global_load_async_to_lds_b128(
            (__attribute__((address_space(1))) gv4i*)(void*)g,
            (__attribute__((address_space(3))) gv4i*)(void*)l, 0, 0);
#else
        *(v8bf*)l = *(const v8bf*)g;
#endif
    }
}

__global__ __launch_bounds__(256)
void gemm_bf16_kernel(const __bf16* __restrict__ A,
                      const __bf16* __restrict__ W,
                      float* __restrict__ C,
                      const float* __restrict__ xres,
                      const unsigned char* __restrict__ am,
                      int M, int N, int K, int mode) {
    __shared__ __bf16 wst[2][GEMM_BN * GEMM_BK];    // 2 x 16 KiB

    const int nbBlocks = N / GEMM_BN;
    const int mb = blockIdx.x / nbBlocks;
    const int nb = blockIdx.x % nbBlocks;
    if (mb * GEMM_BM >= M) return;

    const int tid  = threadIdx.x;
    const int warp = tid >> 5;
    const int lane = tid & 31;
    const int wm = warp >> 2;       // 0..1  (32-row slice)
    const int wn = warp & 3;        // 0..3  (64-col slice)
    const int lh  = lane >> 4;
    const int l15 = lane & 15;

    const int mBase = mb * GEMM_BM;
    const int nBase = nb * GEMM_BN;

    // staging map: thread copies 16B: row = (tid>>2)+64*it, 16B chunk = tid&3
    const int srow   = tid >> 2;
    const int schunk = tid & 3;

    const __bf16* arow0 = A + (size_t)(mBase + wm * 32 + l15) * K;
    const __bf16* arow1 = arow0 + (size_t)16 * K;
    const int kA = 16 * lh;

    v8f acc[2][4] = {};

    // prologue: stage k-slice 0
    gemm_stage_w(&wst[0][0], W, nBase, K, 0, srow, schunk);
#if USE_ASYNC_LDS
    __builtin_amdgcn_s_wait_asynccnt(0);
#endif
    __syncthreads();

    int cur = 0;
    for (int kk = 0; kk < K; kk += GEMM_BK) {
        const int nxt = kk + GEMM_BK;
        if (nxt < K)
            gemm_stage_w(&wst[cur ^ 1][0], W, nBase, K, nxt, srow, schunk);

        // A fragments straight from global (b128-coalesced)
        AFrag a0, a1;
        a0.h[0] = *(const v8bf*)(arow0 + kk + kA);
        a0.h[1] = *(const v8bf*)(arow0 + kk + 8 + kA);
        a1.h[0] = *(const v8bf*)(arow1 + kk + kA);
        a1.h[1] = *(const v8bf*)(arow1 + kk + 8 + kA);

        // B fragments from LDS; consumed by WMMA before the barrier so the
        // compiler's DScnt waits drain all reads of wst[cur] in-iteration.
#pragma unroll
        for (int j = 0; j < 4; ++j) {
            const __bf16* bp =
                &wst[cur][(wn * 64 + j * 16 + l15) * GEMM_BK + lh * 16];
            v16bf b = *(const v16bf*)bp;
            acc[0][j] = wmma_bf16(a0, b, acc[0][j]);
            acc[1][j] = wmma_bf16(a1, b, acc[1][j]);
        }

#if USE_ASYNC_LDS
        __builtin_amdgcn_s_wait_asynccnt(0);   // my next-buffer copies landed
#endif
        __syncthreads();                        // everyone done: reads of cur,
                                                // writes of cur^1
        cur ^= 1;
    }

#pragma unroll
    for (int i = 0; i < 2; ++i)
#pragma unroll
        for (int j = 0; j < 4; ++j)
#pragma unroll
            for (int v = 0; v < 8; ++v) {
                const int r = mBase + wm * 32 + i * 16 + v + 8 * lh;
                const int c = nBase + wn * 64 + j * 16 + l15;
                float val = acc[i][j][v];
                if (mode == 1) {
                    const float keep = am[r] ? 1.f : 0.f;
                    val = xres[(size_t)r * N + c] + keep * val;
                }
                C[(size_t)r * N + c] = val;
            }
}

// ---------------------------------------------------------------------------
// 4) RoPE + repack:  qkv_f32[T][3][H][128] ->
//    q_bf16[H][T][128], k_bf16[H][T][128], vT_bf16[H][128][T]
// ---------------------------------------------------------------------------
__global__ __launch_bounds__(256)
void rope_pack_kernel(const float* __restrict__ qkv,
                      const float* __restrict__ cs,
                      const float* __restrict__ sn,
                      __bf16* __restrict__ q,
                      __bf16* __restrict__ k,
                      __bf16* __restrict__ vT) {
    const int t = blockIdx.x;
    const float* row = qkv + (size_t)t * QKV_N;
    const float* cr = cs + (size_t)t * D_HEAD;
    const float* sr = sn + (size_t)t * D_HEAD;
    for (int idx = threadIdx.x; idx < D_MODEL; idx += blockDim.x) {
        const int hd = idx >> 7;            // head
        const int d  = idx & 127;
        const float c = cr[d], s = sr[d];
        const int   rot = (d < 64) ? d + 64 : d - 64;
        const float sgn = (d < 64) ? -1.f : 1.f;
        const int hb = hd << 7;
        const float qv  = row[idx];
        const float qr  = row[hb + rot];
        const float kv  = row[D_MODEL + idx];
        const float kr  = row[D_MODEL + hb + rot];
        const float vv  = row[2 * D_MODEL + idx];
        q[((size_t)hd * T_SEQ + t) * D_HEAD + d] = (__bf16)(qv * c + sgn * qr * s);
        k[((size_t)hd * T_SEQ + t) * D_HEAD + d] = (__bf16)(kv * c + sgn * kr * s);
        vT[((size_t)hd * D_HEAD + d) * T_SEQ + t] = (__bf16)vv;
    }
}

// ---------------------------------------------------------------------------
// 5) Flash attention: one wave per (head, 16-query tile). Online softmax over
//    32-key tiles; Q.K^T and P.V both via v_wmma_f32_16x16x32_bf16.
//    Single-wave workgroup => per-wave LDS P-transpose is barrier-safe even
//    with causal (divergent-per-wave) KV trip counts.
// ---------------------------------------------------------------------------
__global__ __launch_bounds__(32)
void flash_attn_kernel(const __bf16* __restrict__ q,
                       const __bf16* __restrict__ k,
                       const __bf16* __restrict__ vT,
                       const unsigned char* __restrict__ am,
                       __bf16* __restrict__ attn) {
    __shared__ __bf16 ps[16 * 32];   // P staging (D-layout -> A-layout)

    const int wid  = blockIdx.x;          // 2048 waves: head*128 + qtile
    const int head = wid >> 7;
    const int qt   = wid & 127;
    const int lane = threadIdx.x & 31;
    const int lh   = lane >> 4;
    const int l15  = lane & 15;

    const __bf16* qh = q  + (size_t)head * T_SEQ * D_HEAD;
    const __bf16* kh = k  + (size_t)head * T_SEQ * D_HEAD;
    const __bf16* vh = vT + (size_t)head * D_HEAD * T_SEQ;

    // Preload Q A-fragments for d = 0..127 (4 frags of K=32)
    AFrag af[4];
    const __bf16* qr = qh + (size_t)(qt * 16 + l15) * D_HEAD;
#pragma unroll
    for (int f = 0; f < 4; ++f) {
        af[f].h[0] = *(const v8bf*)(qr + f * 32 + 16 * lh);
        af[f].h[1] = *(const v8bf*)(qr + f * 32 + 8 + 16 * lh);
    }

    float mrow[8], lrow[8];
#pragma unroll
    for (int v = 0; v < 8; ++v) { mrow[v] = -3.0e38f; lrow[v] = 0.f; }
    v8f o[8] = {};

    const int qend = qt * 16 + 15;
    for (int j0 = 0; j0 <= qend; j0 += 32) {
        // prefetch next K tile (global_prefetch_b8) to cover the softmax gap
        if (j0 + 32 <= qend) {
            __builtin_prefetch(kh + (size_t)(j0 + 32 + l15) * D_HEAD, 0, 1);
            __builtin_prefetch(kh + (size_t)(j0 + 48 + l15) * D_HEAD, 0, 1);
        }

        // ---- S = Q.K^T for 32 keys (two 16-col WMMA tiles, K=128) ----
        v8f s0 = {}, s1 = {};
        const __bf16* kr0 = kh + (size_t)(j0 + l15) * D_HEAD + 16 * lh;
        const __bf16* kr1 = kr0 + (size_t)16 * D_HEAD;
#pragma unroll
        for (int f = 0; f < 4; ++f) {
            v16bf b0 = *(const v16bf*)(kr0 + f * 32);
            s0 = wmma_bf16(af[f], b0, s0);
            v16bf b1 = *(const v16bf*)(kr1 + f * 32);
            s1 = wmma_bf16(af[f], b1, s1);
        }

        // ---- mask + scale, online softmax update ----
        const int jc0 = j0 + l15;
        const int jc1 = j0 + 16 + l15;
        const bool am0 = am[jc0] != 0;
        const bool am1 = am[jc1] != 0;
#pragma unroll
        for (int v = 0; v < 8; ++v) {
            const int r = qt * 16 + v + 8 * lh;
            const float sv0 = (am0 && jc0 <= r) ? s0[v] * ATTN_SCALE : NEG_INF;
            const float sv1 = (am1 && jc1 <= r) ? s1[v] * ATTN_SCALE : NEG_INF;
            float tmax = fmaxf(sv0, sv1);
            for (int msk = 8; msk > 0; msk >>= 1)
                tmax = fmaxf(tmax, __shfl_xor(tmax, msk, 32));
            const float mnew  = fmaxf(mrow[v], tmax);
            const float alpha = __expf(mrow[v] - mnew);
            const float p0 = __expf(sv0 - mnew);
            const float p1 = __expf(sv1 - mnew);
            float psum = p0 + p1;
            for (int msk = 8; msk > 0; msk >>= 1)
                psum += __shfl_xor(psum, msk, 32);
            lrow[v] = lrow[v] * alpha + psum;
            mrow[v] = mnew;
#pragma unroll
            for (int nt = 0; nt < 8; ++nt) o[nt][v] *= alpha;
            // stage P (D-layout: row v+8*lh, cols l15 / 16+l15)
            ps[(v + 8 * lh) * 32 + l15]      = (__bf16)p0;
            ps[(v + 8 * lh) * 32 + 16 + l15] = (__bf16)p1;
        }
        __syncthreads();   // single-wave WG: orders ds stores before loads

        // ---- reload P as an A-fragment (16x32, K = 32 keys) ----
        AFrag pa;
        pa.h[0] = *(const v8bf*)(ps + l15 * 32 + 16 * lh);
        pa.h[1] = *(const v8bf*)(ps + l15 * 32 + 8 + 16 * lh);
        __syncthreads();

        // ---- O += P.V  (8 d-tiles of 16; B from vT rows, contiguous) ----
#pragma unroll
        for (int nt = 0; nt < 8; ++nt) {
            const __bf16* vrow =
                vh + (size_t)(nt * 16 + l15) * T_SEQ + j0 + 16 * lh;
            v16bf b = *(const v16bf*)vrow;
            o[nt] = wmma_bf16(pa, b, o[nt]);
        }
    }

    // ---- normalize and write attn_bf16[T][d_model] ----
#pragma unroll
    for (int nt = 0; nt < 8; ++nt)
#pragma unroll
        for (int v = 0; v < 8; ++v) {
            const int r = qt * 16 + v + 8 * lh;
            const int c = head * D_HEAD + nt * 16 + l15;
            const float val = o[nt][v] / fmaxf(lrow[v], 1e-30f);
            attn[(size_t)r * D_MODEL + c] = (__bf16)val;
        }
}

// ---------------------------------------------------------------------------
// kernel_launch
// inputs: 0:x[T,D] 1:cos[T,128] 2:sin[T,128] 3:attention_mask[T](bool)
//         4:ln_w[D] 5:w_qkv[6144,2048] 6:w_o[2048,2048]
// output: f32 [T, D]
// ---------------------------------------------------------------------------
extern "C" void kernel_launch(void* const* d_in, const int* in_sizes, int n_in,
                              void* d_out, int out_size, void* d_ws,
                              size_t ws_size, hipStream_t stream) {
    const float* x    = (const float*)d_in[0];
    const float* cs   = (const float*)d_in[1];
    const float* sn   = (const float*)d_in[2];
    const unsigned char* am = (const unsigned char*)d_in[3];
    const float* ln_w = (const float*)d_in[4];
    const float* wqkv = (const float*)d_in[5];
    const float* wo   = (const float*)d_in[6];
    float* out = (float*)d_out;

    // Workspace layout (bytes); total = 120 MiB
    char* ws = (char*)d_ws;
    const size_t SZ_H    = (size_t)T_SEQ * D_MODEL * 2;   //  8 MiB h bf16
    const size_t SZ_WQKV = (size_t)QKV_N * D_MODEL * 2;   // 24 MiB
    const size_t SZ_WO   = (size_t)D_MODEL * D_MODEL * 2; //  8 MiB
    const size_t SZ_QKV  = (size_t)T_SEQ * QKV_N * 4;     // 48 MiB f32
    const size_t SZ_HEAD = (size_t)N_HEAD * T_SEQ * D_HEAD * 2; // 8 MiB
    __bf16* h_bf    = (__bf16*)(ws);
    __bf16* wqkv_bf = (__bf16*)(ws + SZ_H);
    __bf16* wo_bf   = (__bf16*)(ws + SZ_H + SZ_WQKV);
    float*  qkv_f   = (float*) (ws + SZ_H + SZ_WQKV + SZ_WO);
    __bf16* q_bf    = (__bf16*)(ws + SZ_H + SZ_WQKV + SZ_WO + SZ_QKV);
    __bf16* k_bf    = (__bf16*)(ws + SZ_H + SZ_WQKV + SZ_WO + SZ_QKV + SZ_HEAD);
    __bf16* vT_bf   = (__bf16*)(ws + SZ_H + SZ_WQKV + SZ_WO + SZ_QKV + 2 * SZ_HEAD);
    __bf16* attn_bf = (__bf16*)(ws + SZ_H + SZ_WQKV + SZ_WO + SZ_QKV + 3 * SZ_HEAD);

    // 1) RMSNorm -> bf16 activations
    rmsnorm_bf16_kernel<<<T_SEQ, 256, 0, stream>>>(x, ln_w, h_bf);

    // 2) weight converts
    cvt_bf16_kernel<<<1024, 256, 0, stream>>>(wqkv, wqkv_bf, QKV_N * D_MODEL);
    cvt_bf16_kernel<<<1024, 256, 0, stream>>>(wo, wo_bf, D_MODEL * D_MODEL);

    // 3) QKV GEMM: [2048x2048] @ [6144x2048]^T, block tile 64x256
    {
        const int blocks = (T_SEQ / GEMM_BM) * (QKV_N / GEMM_BN); // 32*24
        gemm_bf16_kernel<<<blocks, 256, 0, stream>>>(
            h_bf, wqkv_bf, qkv_f, nullptr, nullptr,
            T_SEQ, QKV_N, D_MODEL, 0);
    }

    // 4) RoPE + repack (v transposed for contiguous P.V B-fragments)
    rope_pack_kernel<<<T_SEQ, 256, 0, stream>>>(qkv_f, cs, sn, q_bf, k_bf, vT_bf);

    // 5) flash attention: 16 heads * 128 q-tiles = 2048 single-wave blocks
    flash_attn_kernel<<<N_HEAD * (T_SEQ / 16), 32, 0, stream>>>(
        q_bf, k_bf, vT_bf, am, attn_bf);

    // 6) output projection + residual + query mask, block tile 64x256
    {
        const int blocks = (T_SEQ / GEMM_BM) * (D_MODEL / GEMM_BN); // 32*8
        gemm_bf16_kernel<<<blocks, 256, 0, stream>>>(
            attn_bf, wo_bf, out, x, am, T_SEQ, D_MODEL, D_MODEL, 1);
    }
}
